// HashingMemory_28157805592819
// MI455X (gfx1250) — compile-verified, matrix operand order checked by reading, not run on hardware
//
#include <hip/hip_runtime.h>
#include <hip/hip_bf16.h>
#include <math.h>

typedef __bf16 bf16;
typedef __bf16 v16bf __attribute__((ext_vector_type(16)));
typedef float  v8f   __attribute__((ext_vector_type(8)));

#define HEADS 4
#define KNN   32
#define NKEYS 512
#define HALFD 256
#define KDIM  512
#define BS    8192
#define VDIM  512
#define NQ    2048

#define TROW   40                 // padded row stride (bf16 elems): 32 k + 8 pad
#define TILE_E (64 * TROW)        // elems per 64x32 tile
#define TILE_B (TILE_E * 2)       // bytes per tile

// ---------------- fp32 -> bf16 hi/lo split (for bf16x3 WMMA GEMM) -------------
__global__ void split_kernel(const float* __restrict__ src, bf16* __restrict__ hi,
                             bf16* __restrict__ lo, int n) {
  int i = blockIdx.x * blockDim.x + threadIdx.x;
  if (i < n) {
    float x = src[i];
    bf16 h = (bf16)x;
    hi[i] = h;
    lo[i] = (bf16)(x - (float)h);
  }
}

// ---------------- tiled WMMA GEMM: C(MxN) = A(MxK) @ B(NxK)^T ------------------
// bf16 hi/lo inputs, fp32 accum via 3 bf16 WMMA products. 64x64 block tile,
// 8 waves (4 M-sub x 2 N-sub). Global->LDS via async-to-LDS, double buffered.
// mode 0: fp32 out (+bias); mode 1: silu(fp32+bias); mode 2: bf16 hi/lo out.
union Frag { uint4 u[2]; v16bf v; };

typedef __attribute__((address_space(3))) const bf16 lds_cbf16;

__device__ __forceinline__ void async_tile16(unsigned lds_off, const bf16* gp) {
  // GLOBAL_LOAD_ASYNC_TO_LDS_B128: vdst = LDS byte offset, vaddr = 64-bit addr
  asm volatile("global_load_async_to_lds_b128 %0, %1, off"
               :: "v"(lds_off), "v"(gp)
               : "memory");
}

__global__ __launch_bounds__(256) void gemm_bf16x3(
    const bf16* __restrict__ Ahi, const bf16* __restrict__ Alo, int lda,
    const bf16* __restrict__ Bhi, const bf16* __restrict__ Blo, int ldb,
    float* __restrict__ C, bf16* __restrict__ Chi, bf16* __restrict__ Clo, int ldc,
    const float* __restrict__ bias, int K, int mode) {
  // double buffer x 4 tiles (Ah, Al, Bh, Bl)
  __shared__ __align__(16) bf16 smem[2 * 4 * TILE_E];

  const int t    = threadIdx.x;
  const int lane = t & 31;
  const int wid  = t >> 5;
  const int wm   = wid & 3;     // M sub-tile (16 rows)
  const int wn   = wid >> 2;    // N sub-tile (32 cols -> two 16x16 accums)
  const int bm   = blockIdx.y, bn = blockIdx.x;

  const int lrow = t >> 2;        // 0..63
  const int lseg = (t & 3) * 8;   // 0,8,16,24 (8 bf16 = 16B per thread)
  const size_t aOff = (size_t)(bm * 64 + lrow) * (size_t)lda + lseg;
  const size_t bOff = (size_t)(bn * 64 + lrow) * (size_t)ldb + lseg;
  const bf16* gAh = Ahi + aOff;
  const bf16* gAl = Alo + aOff;
  const bf16* gBh = Bhi + bOff;
  const bf16* gBl = Blo + bOff;

  // LDS byte offset of smem base (addrspace(3) ptrtoint)
  const unsigned smbase = (unsigned)(unsigned long long)(lds_cbf16*)smem;
  const unsigned wrOff  = smbase + (unsigned)((lrow * TROW + lseg) * 2);

  // WMMA operand lane mapping (16-bit A/B 16x32): row = lane%16,
  // k in {koff..koff+7} U {koff+16..koff+23}, koff = (lane/16)*8.
  const int mrow = wm * 16 + (lane & 15);
  const int koff = (lane >> 4) * 8;
  const int nc0  = wn * 32 + (lane & 15);
  const int nc1  = nc0 + 16;

  v8f acc0 = {}; v8f acc1 = {};

  const int nk = K >> 5;
  // prologue: async-copy tile 0 into buffer 0
  {
    const unsigned b = wrOff;
    async_tile16(b + 0u * TILE_B, gAh);
    async_tile16(b + 1u * TILE_B, gAl);
    async_tile16(b + 2u * TILE_B, gBh);
    async_tile16(b + 3u * TILE_B, gBl);
  }

  for (int kt = 0; kt < nk; ++kt) {
    const int cur = kt & 1;
    if (kt + 1 < nk) {
      // issue next tile into the other buffer, then wait for current tile
      const int k0 = (kt + 1) * 32;
      const unsigned b = wrOff + (unsigned)((1 - cur) * 4) * TILE_B;
      async_tile16(b + 0u * TILE_B, gAh + k0);
      async_tile16(b + 1u * TILE_B, gAl + k0);
      async_tile16(b + 2u * TILE_B, gBh + k0);
      async_tile16(b + 3u * TILE_B, gBl + k0);
      asm volatile("s_wait_asynccnt 0x4" ::: "memory");
    } else {
      asm volatile("s_wait_asynccnt 0x0" ::: "memory");
    }
    __syncthreads();  // current buffer visible to all waves

    const bf16* tb  = smem + cur * 4 * TILE_E;
    const bf16* tAh = tb + 0 * TILE_E;
    const bf16* tAl = tb + 1 * TILE_E;
    const bf16* tBh = tb + 2 * TILE_E;
    const bf16* tBl = tb + 3 * TILE_E;

    Frag ah, al, b0h, b0l, b1h, b1l;
    ah.u[0]  = *(const uint4*)(tAh + mrow * TROW + koff);
    ah.u[1]  = *(const uint4*)(tAh + mrow * TROW + koff + 16);
    al.u[0]  = *(const uint4*)(tAl + mrow * TROW + koff);
    al.u[1]  = *(const uint4*)(tAl + mrow * TROW + koff + 16);
    b0h.u[0] = *(const uint4*)(tBh + nc0 * TROW + koff);
    b0h.u[1] = *(const uint4*)(tBh + nc0 * TROW + koff + 16);
    b0l.u[0] = *(const uint4*)(tBl + nc0 * TROW + koff);
    b0l.u[1] = *(const uint4*)(tBl + nc0 * TROW + koff + 16);
    b1h.u[0] = *(const uint4*)(tBh + nc1 * TROW + koff);
    b1h.u[1] = *(const uint4*)(tBh + nc1 * TROW + koff + 16);
    b1l.u[0] = *(const uint4*)(tBl + nc1 * TROW + koff);
    b1l.u[1] = *(const uint4*)(tBl + nc1 * TROW + koff + 16);

    // bf16x3 emulated-fp32 products
    acc0 = __builtin_amdgcn_wmma_f32_16x16x32_bf16(false, ah.v, false, b0h.v, (short)0, acc0, false, false);
    acc0 = __builtin_amdgcn_wmma_f32_16x16x32_bf16(false, ah.v, false, b0l.v, (short)0, acc0, false, false);
    acc0 = __builtin_amdgcn_wmma_f32_16x16x32_bf16(false, al.v, false, b0h.v, (short)0, acc0, false, false);
    acc1 = __builtin_amdgcn_wmma_f32_16x16x32_bf16(false, ah.v, false, b1h.v, (short)0, acc1, false, false);
    acc1 = __builtin_amdgcn_wmma_f32_16x16x32_bf16(false, ah.v, false, b1l.v, (short)0, acc1, false, false);
    acc1 = __builtin_amdgcn_wmma_f32_16x16x32_bf16(false, al.v, false, b1h.v, (short)0, acc1, false, false);
    __syncthreads();  // all reads of current buffer done before it is reused
  }

  // epilogue; C/D layout: n = lane%16, m = vgpr + 8*(lane/16)
  const int gn0 = bn * 64 + nc0;
  const int gn1 = bn * 64 + nc1;
  const float bia0 = bias ? bias[gn0] : 0.0f;
  const float bia1 = bias ? bias[gn1] : 0.0f;
  const int mbase = bm * 64 + wm * 16 + (lane >> 4) * 8;
#pragma unroll
  for (int v = 0; v < 8; ++v) {
    const int m = mbase + v;
    float c0 = acc0[v] + bia0;
    float c1 = acc1[v] + bia1;
    if (mode == 1) {  // silu
      c0 = c0 / (1.0f + __expf(-c0));
      c1 = c1 / (1.0f + __expf(-c1));
    }
    if (mode == 2) {  // bf16 hi/lo split output
      bf16 h0 = (bf16)c0;
      Chi[(size_t)m * ldc + gn0] = h0;
      Clo[(size_t)m * ldc + gn0] = (bf16)(c0 - (float)h0);
      bf16 h1 = (bf16)c1;
      Chi[(size_t)m * ldc + gn1] = h1;
      Clo[(size_t)m * ldc + gn1] = (bf16)(c1 - (float)h1);
    } else {
      C[(size_t)m * ldc + gn0] = c0;
      C[(size_t)m * ldc + gn1] = c1;
    }
  }
}

// ---------------- wave32 top-32-of-512 (one wave per row) ----------------------
__global__ __launch_bounds__(256) void topk32_kernel(const float* __restrict__ s,
                                                     float* __restrict__ outv,
                                                     int* __restrict__ outi) {
  const int lane = threadIdx.x & 31;
  const int row  = blockIdx.x * 8 + (threadIdx.x >> 5);
  const float* sr = s + (size_t)row * NKEYS;
  float v[16];
#pragma unroll
  for (int i = 0; i < 16; ++i) v[i] = sr[lane + i * 32];
  for (int t = 0; t < KNN; ++t) {
    float mv = v[0]; int mi = 0;
#pragma unroll
    for (int i = 1; i < 16; ++i) if (v[i] > mv) { mv = v[i]; mi = i; }
    int midx = mi * 32 + lane;  // flat element index == memory index
    for (int off = 16; off > 0; off >>= 1) {
      float ov = __shfl_xor(mv, off, 32);
      int   oi = __shfl_xor(midx, off, 32);
      if (ov > mv || (ov == mv && oi < midx)) { mv = ov; midx = oi; }
    }
    if (lane == 0) { outv[(size_t)row * KNN + t] = mv; outi[(size_t)row * KNN + t] = midx; }
    if ((midx & 31) == lane) v[midx >> 5] = -__builtin_inff();
  }
}

// ------------- cartesian 32x32 top-32 + softmax (one wave per row*head) --------
__global__ __launch_bounds__(256) void combine_kernel(const float* __restrict__ scb,
                                                      const int* __restrict__ icb,
                                                      float* __restrict__ wout,
                                                      int* __restrict__ iout) {
  const int lane = threadIdx.x & 31;
  const int g    = blockIdx.x * 8 + (threadIdx.x >> 5);
  const int row  = g >> 2;
  const int h    = g & 3;
  const size_t o1 = ((size_t)(0 * HEADS + h) * BS + row) * KNN + lane;
  const size_t o2 = ((size_t)(1 * HEADS + h) * BS + row) * KNN + lane;
  const float sc1 = scb[o1]; const int i1 = icb[o1];
  const float sc2 = scb[o2]; const int i2 = icb[o2];
  float cand[32];
#pragma unroll
  for (int b = 0; b < 32; ++b) cand[b] = sc1 + __shfl(sc2, b, 32);
  float myscore = 0.0f; int myindex = 0;
  for (int t = 0; t < KNN; ++t) {
    float mv = cand[0]; int mb = 0;
#pragma unroll
    for (int b = 1; b < 32; ++b) if (cand[b] > mv) { mv = cand[b]; mb = b; }
    int flat = lane * 32 + mb;  // i*KNN + j as in reference
    for (int off = 16; off > 0; off >>= 1) {
      float ov = __shfl_xor(mv, off, 32);
      int   of = __shfl_xor(flat, off, 32);
      if (ov > mv || (ov == mv && of < flat)) { mv = ov; flat = of; }
    }
    const int id1 = __shfl(i1, flat >> 5, 32);
    const int id2 = __shfl(i2, flat & 31, 32);
    if (lane == t) { myscore = mv; myindex = id1 * NKEYS + id2; }
    if ((flat >> 5) == lane) cand[flat & 31] = -__builtin_inff();
  }
  // softmax over the 32 selected scores (lane 0 holds the max)
  const float mx = __shfl(myscore, 0, 32);
  float e = __expf(myscore - mx);
  float ssum = e;
  for (int off = 16; off > 0; off >>= 1) ssum += __shfl_xor(ssum, off, 32);
  wout[(size_t)row * (HEADS * KNN) + h * KNN + lane] = e / ssum;
  iout[(size_t)row * (HEADS * KNN) + h * KNN + lane] = myindex;
}

// -------- embedding-bag gather + SiLU gate multiply + bf16 hi/lo split ---------
__global__ __launch_bounds__(128) void bag_kernel(const float* __restrict__ w,
                                                  const int* __restrict__ idx,
                                                  const float* __restrict__ values,
                                                  const float* __restrict__ gate,
                                                  bf16* __restrict__ fhi,
                                                  bf16* __restrict__ flo) {
  __shared__ float sw[HEADS * KNN];
  __shared__ int   si[HEADS * KNN];
  const int t = threadIdx.x, row = blockIdx.x;
  sw[t] = w[(size_t)row * (HEADS * KNN) + t];
  si[t] = idx[(size_t)row * (HEADS * KNN) + t];
  __syncthreads();
  float a0 = 0.f, a1 = 0.f, a2 = 0.f, a3 = 0.f;
  const float4* V = reinterpret_cast<const float4*>(values);
  for (int k = 0; k < HEADS * KNN; ++k) {
    if (k + 8 < HEADS * KNN)  // prefetch a future gathered row (global_prefetch_b8)
      __builtin_prefetch(&V[(size_t)si[k + 8] * (VDIM / 4) + t], 0, 0);
    const float wk = sw[k];
    const float4 vv = V[(size_t)si[k] * (VDIM / 4) + t];
    a0 += wk * vv.x; a1 += wk * vv.y; a2 += wk * vv.z; a3 += wk * vv.w;
  }
  const float4 gg = reinterpret_cast<const float4*>(gate)[(size_t)row * (VDIM / 4) + t];
  const float f[4] = { a0 * gg.x, a1 * gg.y, a2 * gg.z, a3 * gg.w };
  const size_t base = (size_t)row * VDIM + (size_t)t * 4;
#pragma unroll
  for (int c = 0; c < 4; ++c) {
    bf16 hh = (bf16)f[c];
    fhi[base + c] = hh;
    flo[base + c] = (bf16)(f[c] - (float)hh);
  }
}

// ------------------------------- host driver ----------------------------------
extern "C" void kernel_launch(void* const* d_in, const int* in_sizes, int n_in,
                              void* d_out, int out_size, void* d_ws, size_t ws_size,
                              hipStream_t stream) {
  (void)in_sizes; (void)n_in; (void)out_size; (void)ws_size;
  const float* x      = (const float*)d_in[0];
  const float* W_q    = (const float*)d_in[1];
  const float* b_q    = (const float*)d_in[2];
  const float* keys   = (const float*)d_in[3];
  const float* values = (const float*)d_in[4];
  const float* W_sw   = (const float*)d_in[5];
  const float* b_sw   = (const float*)d_in[6];
  const float* W_vp   = (const float*)d_in[7];
  const float* b_vp   = (const float*)d_in[8];
  float* out = (float*)d_out;

  char* ws = (char*)d_ws;
  size_t off = 0;
  auto alloc = [&](size_t bytes) -> char* {
    char* p = ws + off;
    off = (off + bytes + 255) & ~(size_t)255;
    return p;
  };

  bf16*  xhi   = (bf16*)alloc((size_t)BS * KDIM * 2);
  bf16*  xlo   = (bf16*)alloc((size_t)BS * KDIM * 2);
  bf16*  wqhi  = (bf16*)alloc((size_t)NQ * KDIM * 2);
  bf16*  wqlo  = (bf16*)alloc((size_t)NQ * KDIM * 2);
  bf16*  khi   = (bf16*)alloc((size_t)2 * HEADS * NKEYS * HALFD * 2);
  bf16*  klo   = (bf16*)alloc((size_t)2 * HEADS * NKEYS * HALFD * 2);
  bf16*  wswhi = (bf16*)alloc((size_t)VDIM * KDIM * 2);
  bf16*  wswlo = (bf16*)alloc((size_t)VDIM * KDIM * 2);
  bf16*  wvphi = (bf16*)alloc((size_t)VDIM * VDIM * 2);
  bf16*  wvplo = (bf16*)alloc((size_t)VDIM * VDIM * 2);
  bf16*  qhi   = (bf16*)alloc((size_t)BS * NQ * 2);
  bf16*  qlo   = (bf16*)alloc((size_t)BS * NQ * 2);
  float* gate  = (float*)alloc((size_t)BS * VDIM * 4);
  float* sbuf  = (float*)alloc((size_t)BS * NKEYS * 4);           // reused per (set,head)
  float* scb   = (float*)alloc((size_t)2 * HEADS * BS * KNN * 4);
  int*   icb   = (int*)  alloc((size_t)2 * HEADS * BS * KNN * 4);
  float* wbuf  = (float*)alloc((size_t)BS * HEADS * KNN * 4);
  int*   ibuf  = (int*)  alloc((size_t)BS * HEADS * KNN * 4);
  bf16*  fhi   = (bf16*)alloc((size_t)BS * VDIM * 2);
  bf16*  flo   = (bf16*)alloc((size_t)BS * VDIM * 2);

  // 1) split inputs into bf16 hi/lo
  auto split = [&](const float* src, bf16* h, bf16* l, int n) {
    split_kernel<<<(n + 255) / 256, 256, 0, stream>>>(src, h, l, n);
  };
  split(x,    xhi,   xlo,   BS * KDIM);
  split(W_q,  wqhi,  wqlo,  NQ * KDIM);
  split(keys, khi,   klo,   2 * HEADS * NKEYS * HALFD);
  split(W_sw, wswhi, wswlo, VDIM * KDIM);
  split(W_vp, wvphi, wvplo, VDIM * VDIM);

  // 2) q = x @ W_q^T + b_q  -> bf16 hi/lo split output (mode 2)
  gemm_bf16x3<<<dim3(NQ / 64, BS / 64), 256, 0, stream>>>(
      xhi, xlo, KDIM, wqhi, wqlo, KDIM,
      nullptr, qhi, qlo, NQ, b_q, KDIM, 2);

  // 3) gate = silu(x @ W_sw^T + b_sw)  (mode 1)
  gemm_bf16x3<<<dim3(VDIM / 64, BS / 64), 256, 0, stream>>>(
      xhi, xlo, KDIM, wswhi, wswlo, KDIM,
      gate, nullptr, nullptr, VDIM, b_sw, KDIM, 1);

  // 4) per (set, head): scores GEMM then wave top-32 (sbuf reused serially)
  for (int j = 0; j < 2; ++j) {
    for (int h = 0; h < HEADS; ++h) {
      const bf16* aH = qhi + h * KDIM + j * HALFD;
      const bf16* aL = qlo + h * KDIM + j * HALFD;
      const bf16* bH = khi + (size_t)((j * HEADS + h) * NKEYS) * HALFD;
      const bf16* bL = klo + (size_t)((j * HEADS + h) * NKEYS) * HALFD;
      gemm_bf16x3<<<dim3(NKEYS / 64, BS / 64), 256, 0, stream>>>(
          aH, aL, NQ, bH, bL, HALFD,
          sbuf, nullptr, nullptr, NKEYS, nullptr, HALFD, 0);
      topk32_kernel<<<BS / 8, 256, 0, stream>>>(
          sbuf,
          scb + (size_t)(j * HEADS + h) * BS * KNN,
          icb + (size_t)(j * HEADS + h) * BS * KNN);
    }
  }

  // 5) cartesian top-32 + softmax
  combine_kernel<<<(BS * HEADS) / 8, 256, 0, stream>>>(scb, icb, wbuf, ibuf);

  // 6) embedding bag gather * gate -> bf16 hi/lo
  bag_kernel<<<BS, 128, 0, stream>>>(wbuf, ibuf, values, gate, fhi, flo);

  // 7) out = fused @ W_vp^T + b_vp
  gemm_bf16x3<<<dim3(VDIM / 64, BS / 64), 256, 0, stream>>>(
      fhi, flo, VDIM, wvphi, wvplo, VDIM,
      out, nullptr, nullptr, VDIM, b_vp, KDIM, 0);
}
